// MaskedMultiHeadAttention_11862699671873
// MI455X (gfx1250) — compile-verified, matrix operand order checked by reading, not run on hardware
//
#include <hip/hip_runtime.h>
#include <math.h>

// ---- CDNA5 vector types ----
typedef __attribute__((ext_vector_type(16))) __bf16 v16bf;
typedef __attribute__((ext_vector_type(8)))  __bf16 v8bf;
typedef __attribute__((ext_vector_type(4)))  __bf16 v4bf;
typedef __attribute__((ext_vector_type(8)))  float  v8f;
typedef __attribute__((ext_vector_type(4)))  unsigned int u32x4;
typedef __attribute__((ext_vector_type(8)))  int          i32x8;
typedef __attribute__((ext_vector_type(4)))  int          i32x4;

// Problem constants (from reference): B=4, S=4096, D=512, TEMP=8
static constexpr int  Bn = 4;
static constexpr int  Sn = 4096;
static constexpr int  Dn = 512;

// Tensor Data Mover availability. This toolchain's builtin takes 6 args
// (clang-23 / therock-10.0 form): (u32x4, i32x8, i32x4, i32x4, i32x8, i32).
#if defined(__HIP_DEVICE_COMPILE__) && \
    __has_builtin(__builtin_amdgcn_tensor_load_to_lds) && \
    __has_builtin(__builtin_amdgcn_s_wait_tensorcnt)
#define USE_TDM 1
#else
#define USE_TDM 0
#endif

// -------------------------------------------------------------------------
// fp32 -> bf16 elementwise convert (for the weight matrices)
// -------------------------------------------------------------------------
__global__ __launch_bounds__(256) void f32_to_bf16_vec4(
    const float* __restrict__ src, __bf16* __restrict__ dst, int n4)
{
    int i = blockIdx.x * blockDim.x + threadIdx.x;
    if (i < n4) {
        float4 f = ((const float4*)src)[i];
        v4bf t;
        t[0] = (__bf16)f.x; t[1] = (__bf16)f.y;
        t[2] = (__bf16)f.z; t[3] = (__bf16)f.w;
        ((v4bf*)dst)[i] = t;
    }
}

#if USE_TDM
// -------------------------------------------------------------------------
// Issue one TDM 2D tile load: `rows` rows of 32 bf16 (64B) with global row
// stride `ldElems` (in elements), into LDS at `ldsByteAddr` with HW padding
// of 4 DWORDs after every 16 DWORDs -> padded LDS row stride of 40 bf16,
// matching the [rows][40] __shared__ arrays below.
// Descriptor per CDNA5 ISA 8.3/8.4: count=1, type=2 (image), data_size=2B,
// tensor dims == tile dims (no OOB), tensor_dim0_stride = ldElems.
// -------------------------------------------------------------------------
__device__ __forceinline__ void tdm_load_tile_b32x(
    const __bf16* gbase, unsigned ldsByteAddr, int rows, int ldElems)
{
    unsigned long long ga = (unsigned long long)(uintptr_t)gbase;
    u32x4 g0;
    g0[0] = 1u;                                        // count=1, user mode
    g0[1] = ldsByteAddr;                               // lds_addr
    g0[2] = (unsigned)(ga & 0xFFFFFFFFu);              // global_addr[31:0]
    g0[3] = (unsigned)((ga >> 32) & 0x01FFFFFFu)       // global_addr[56:32]
          | (2u << 30);                                // type = 2
    i32x8 g1;
    g1[0] = (int)((1u << 16)                           // data_size = 2 bytes
                | (1u << 20)                           // pad_enable
                | (3u << 22)                           // pad_interval: 16 DW
                | (3u << 25));                         // pad_amount: 4 DW
    g1[1] = (32 << 16);                                // tensor_dim0 = 32
    g1[2] = (rows & 0xFFFF) << 16;                     // tensor_dim1 = rows
    g1[3] = (32 << 16);                                // tile_dim0 = 32
    g1[4] = rows & 0xFFFF;                             // tile_dim1 = rows
    g1[5] = ldElems;                                   // tensor_dim0_stride lo
    g1[6] = 0;                                         // stride hi / dim1_stride lo
    g1[7] = 0;
    i32x4 z4 = {0, 0, 0, 0};                           // groups 2/3 unused (2D)
    i32x8 z8 = {0, 0, 0, 0, 0, 0, 0, 0};               // extra group unused
    __builtin_amdgcn_tensor_load_to_lds(g0, g1, z4, z4, z8, 0);
}
#endif

// -------------------------------------------------------------------------
// Tiled bf16 WMMA GEMM:  C[m,n] = alpha * sum_k A[m,k] * Bt[n,k]
//   A  : row-major [M,K], fp32 (converted on stage) or bf16
//   Bt : row-major [N,K] bf16  (computes A @ B with B given as B^T)
//   C  : [M,N] (or [N,M] if STORE_T), fp32 or bf16
// Block: 256 threads = 8 waves (2x4); block tile 64(M) x 256(N); K-step 32.
// Each wave owns a 32x64 C tile = 8 x v_wmma_f32_16x16x32_bf16 per K-step.
// B (and A when bf16) tiles are staged by the Tensor Data Mover.
// -------------------------------------------------------------------------
template<bool A_F32, bool OUT_F32, bool STORE_T>
__global__ __launch_bounds__(256) void gemm_bf16_wmma(
    const void* __restrict__ Av, const __bf16* __restrict__ Bt,
    void* __restrict__ Cv,
    int M, int N, int K, int lda, int ldb, int ldc,
    long long aStride, long long bStride, long long cStride, float alpha)
{
    // padded rows (40 bf16 = 80B) to spread LDS bank accesses; the TDM pad
    // config reproduces exactly this stride.
    __shared__ __bf16 As[64][40];
    __shared__ __bf16 Bs[256][40];

    const int tid  = threadIdx.x;
    const int lane = tid & 31;
    const int wave = tid >> 5;
    const int wr   = wave >> 2;   // 0..1 -> 32-row strip of block tile
    const int wc   = wave & 3;    // 0..3 -> 64-col strip of block tile
    const int rowM = blockIdx.y * 64;
    const int colN = blockIdx.x * 256;
    const int bz   = blockIdx.z;

    const float*  Af = (const float*)Av + (A_F32 ? (size_t)bz * aStride : 0);
    const __bf16* Ab = (const __bf16*)Av + (A_F32 ? 0 : (size_t)bz * aStride);
    const __bf16* Bp = Bt + (size_t)bz * bStride;
    float*  Cf = (float*)Cv  + (size_t)bz * cStride;
    __bf16* Cb = (__bf16*)Cv + (size_t)bz * cStride;

    v8f acc[2][4] = {};   // fp32 accumulators, 8 x 16x16 tiles per wave

    const int sRow = tid >> 2;        // 0..63
    const int sCol = (tid & 3) * 8;   // 0,8,16,24

#if USE_TDM
    const unsigned ldsA = (unsigned)(uintptr_t)&As[0][0];
    const unsigned ldsB = (unsigned)(uintptr_t)&Bs[0][0];
#endif

    for (int kt = 0; kt < K; kt += 32) {
        __syncthreads();   // previous tile fully consumed before overwrite
#if USE_TDM
        if (wave == 0) {
            tdm_load_tile_b32x(Bp + (size_t)colN * ldb + kt, ldsB, 256, ldb);
            if constexpr (!A_F32)
                tdm_load_tile_b32x(Ab + (size_t)rowM * lda + kt, ldsA, 64, lda);
            __builtin_amdgcn_s_wait_tensorcnt(0);  // DMA done before signal
        }
        if constexpr (A_F32) {
            const float* ap = Af + (size_t)(rowM + sRow) * lda + kt + sCol;
            float4 f0 = *(const float4*)ap;
            float4 f1 = *(const float4*)(ap + 4);
            v8bf t;
            t[0] = (__bf16)f0.x; t[1] = (__bf16)f0.y;
            t[2] = (__bf16)f0.z; t[3] = (__bf16)f0.w;
            t[4] = (__bf16)f1.x; t[5] = (__bf16)f1.y;
            t[6] = (__bf16)f1.z; t[7] = (__bf16)f1.w;
            *(v8bf*)&As[sRow][sCol] = t;
            if (kt + 32 < K) __builtin_prefetch(ap + 32, 0, 1);
        }
#else
        // ---- manual fallback staging ----
        if constexpr (A_F32) {
            const float* ap = Af + (size_t)(rowM + sRow) * lda + kt + sCol;
            float4 f0 = *(const float4*)ap;
            float4 f1 = *(const float4*)(ap + 4);
            v8bf t;
            t[0] = (__bf16)f0.x; t[1] = (__bf16)f0.y;
            t[2] = (__bf16)f0.z; t[3] = (__bf16)f0.w;
            t[4] = (__bf16)f1.x; t[5] = (__bf16)f1.y;
            t[6] = (__bf16)f1.z; t[7] = (__bf16)f1.w;
            *(v8bf*)&As[sRow][sCol] = t;
        } else {
            const __bf16* ap = Ab + (size_t)(rowM + sRow) * lda + kt + sCol;
            *(v8bf*)&As[sRow][sCol] = *(const v8bf*)ap;
        }
        {
            const __bf16* bp = Bp + (size_t)(colN + sRow) * ldb + kt + sCol;
            #pragma unroll
            for (int rr = 0; rr < 256; rr += 64)
                *(v8bf*)&Bs[sRow + rr][sCol] =
                    *(const v8bf*)(bp + (size_t)rr * ldb);
        }
#endif
        __syncthreads();

        // ---- per-lane fragment gather, ISA bf16 16x32 layout:
        // lanes 0-15 : row = lane,     elems = K 0..7  and 16..23
        // lanes 16-31: row = lane-16,  elems = K 8..15 and 24..31
        const int frow = lane & 15;
        const int fcol = (lane >> 4) * 8;
        union U { v16bf v; v8bf h[2]; };
        U a[2], b[4];
        #pragma unroll
        for (int mt = 0; mt < 2; ++mt) {
            a[mt].h[0] = *(const v8bf*)&As[wr * 32 + mt * 16 + frow][fcol];
            a[mt].h[1] = *(const v8bf*)&As[wr * 32 + mt * 16 + frow][fcol + 16];
        }
        #pragma unroll
        for (int nt = 0; nt < 4; ++nt) {
            b[nt].h[0] = *(const v8bf*)&Bs[wc * 64 + nt * 16 + frow][fcol];
            b[nt].h[1] = *(const v8bf*)&Bs[wc * 64 + nt * 16 + frow][fcol + 16];
        }
        #pragma unroll
        for (int mt = 0; mt < 2; ++mt)
            #pragma unroll
            for (int nt = 0; nt < 4; ++nt)
                acc[mt][nt] = __builtin_amdgcn_wmma_f32_16x16x32_bf16(
                    false, a[mt].v, false, b[nt].v,
                    (short)0, acc[mt][nt], false, false);
    }

    // ---- store C: VGPR r holds M = r (lanes 0-15) / r+8 (lanes 16-31)
    const int mBase = rowM + wr * 32 + ((lane >> 4) * 8);
    const int nBase = colN + wc * 64 + (lane & 15);
    #pragma unroll
    for (int mt = 0; mt < 2; ++mt)
        #pragma unroll
        for (int nt = 0; nt < 4; ++nt)
            #pragma unroll
            for (int r = 0; r < 8; ++r) {
                const int m = mBase + mt * 16 + r;
                const int n = nBase + nt * 16;
                const float val = acc[mt][nt][r] * alpha;
                if constexpr (STORE_T) {
                    if constexpr (OUT_F32) Cf[(size_t)n * ldc + m] = val;
                    else                   Cb[(size_t)n * ldc + m] = (__bf16)val;
                } else {
                    if constexpr (OUT_F32) Cf[(size_t)m * ldc + n] = val;
                    else                   Cb[(size_t)m * ldc + n] = (__bf16)val;
                }
            }
}

// -------------------------------------------------------------------------
// In-place row softmax over 4096 columns. One 256-thread block per row.
// -------------------------------------------------------------------------
__global__ __launch_bounds__(256) void softmax_rows(float* __restrict__ attn)
{
    constexpr int COLS = Sn;          // 4096
    constexpr int PER  = COLS / 256;  // 16
    float* p = attn + (size_t)blockIdx.x * COLS;
    const int tid = threadIdx.x;
    __shared__ float red[256];

    float vals[PER];
    float mx = -3.402823466e38f;
    #pragma unroll
    for (int j = 0; j < PER; ++j) {
        vals[j] = p[tid + j * 256];
        mx = fmaxf(mx, vals[j]);
    }
    red[tid] = mx;
    __syncthreads();
    for (int s = 128; s > 0; s >>= 1) {
        if (tid < s) red[tid] = fmaxf(red[tid], red[tid + s]);
        __syncthreads();
    }
    mx = red[0];
    __syncthreads();

    float sum = 0.f;
    #pragma unroll
    for (int j = 0; j < PER; ++j) {
        vals[j] = __expf(vals[j] - mx);
        sum += vals[j];
    }
    red[tid] = sum;
    __syncthreads();
    for (int s = 128; s > 0; s >>= 1) {
        if (tid < s) red[tid] += red[tid + s];
        __syncthreads();
    }
    const float inv = 1.0f / red[0];
    #pragma unroll
    for (int j = 0; j < PER; ++j) p[tid + j * 256] = vals[j] * inv;
}

// -------------------------------------------------------------------------
// Launch: ws layout (bytes):
//   [0,16M)   qb  bf16 [B*S, D]   (reused as x_attn after scores)
//   [16,32M)  kb  bf16 [B*S, D]
//   [32,48M)  vT  bf16 per-batch [D, S]
//   [48M,+2M) WQb,WKb,WVb,WOb bf16 [D,D] each
// -------------------------------------------------------------------------
extern "C" void kernel_launch(void* const* d_in, const int* in_sizes, int n_in,
                              void* d_out, int out_size, void* d_ws, size_t ws_size,
                              hipStream_t stream)
{
    (void)in_sizes; (void)n_in; (void)out_size; (void)ws_size;

    const float* Qp = (const float*)d_in[0];
    const float* Kp = (const float*)d_in[1];
    const float* Vp = (const float*)d_in[2];
    const float* WQ = (const float*)d_in[3];
    const float* WK = (const float*)d_in[4];
    const float* WV = (const float*)d_in[5];
    const float* WO = (const float*)d_in[6];

    float* out_x    = (float*)d_out;                       // [B,S,D]
    float* out_attn = out_x + (size_t)Bn * Sn * Dn;        // [B,S,S]

    char*   ws  = (char*)d_ws;
    __bf16* qb  = (__bf16*)(ws);
    __bf16* kb  = (__bf16*)(ws + (size_t)16 * 1024 * 1024);
    __bf16* vT  = (__bf16*)(ws + (size_t)32 * 1024 * 1024);
    __bf16* WQb = (__bf16*)(ws + (size_t)48 * 1024 * 1024);
    __bf16* WKb = WQb + Dn * Dn;
    __bf16* WVb = WKb + Dn * Dn;
    __bf16* WOb = WVb + Dn * Dn;
    __bf16* xb  = qb;   // q no longer needed after scores -> reuse

    const dim3 blk(256);

    // 1) weights -> bf16  (D*D/4 = 65536 float4 per matrix)
    f32_to_bf16_vec4<<<256, blk, 0, stream>>>(WQ, WQb, Dn * Dn / 4);
    f32_to_bf16_vec4<<<256, blk, 0, stream>>>(WK, WKb, Dn * Dn / 4);
    f32_to_bf16_vec4<<<256, blk, 0, stream>>>(WV, WVb, Dn * Dn / 4);
    f32_to_bf16_vec4<<<256, blk, 0, stream>>>(WO, WOb, Dn * Dn / 4);

    // 2) q = Q @ WQ^T, k = K @ WK^T  (M = B*S, single "batch")
    const dim3 gProj(Dn / 256, (Bn * Sn) / 64, 1);
    gemm_bf16_wmma<true, false, false><<<gProj, blk, 0, stream>>>(
        Qp, WQb, qb, Bn * Sn, Dn, Dn, Dn, Dn, Dn, 0, 0, 0, 1.0f);
    gemm_bf16_wmma<true, false, false><<<gProj, blk, 0, stream>>>(
        Kp, WKb, kb, Bn * Sn, Dn, Dn, Dn, Dn, Dn, 0, 0, 0, 1.0f);

    // 3) v = V @ WV^T stored transposed per batch: vT[b][e][s]
    const dim3 gProjV(Dn / 256, Sn / 64, Bn);
    gemm_bf16_wmma<true, false, true><<<gProjV, blk, 0, stream>>>(
        Vp, WVb, vT, Sn, Dn, Dn, Dn, Dn, Sn,
        (long long)Sn * Dn, 0, (long long)Dn * Sn, 1.0f);

    // 4) scores = (q @ k^T) / 8  -> fp32 directly into d_out attn region
    const dim3 gSc(Sn / 256, Sn / 64, Bn);
    gemm_bf16_wmma<false, true, false><<<gSc, blk, 0, stream>>>(
        qb, kb, out_attn, Sn, Sn, Dn, Dn, Dn, Sn,
        (long long)Sn * Dn, (long long)Sn * Dn, (long long)Sn * Sn, 0.125f);

    // 5) softmax rows in place (B*S rows of 4096)
    softmax_rows<<<Bn * Sn, blk, 0, stream>>>(out_attn);

    // 6) x_attn = attn @ v   (A fp32 from d_out, B^T = vT, out bf16)
    const dim3 gAV(Dn / 256, Sn / 64, Bn);
    gemm_bf16_wmma<true, false, false><<<gAV, blk, 0, stream>>>(
        out_attn, vT, xb, Sn, Dn, Sn, Sn, Sn, Dn,
        (long long)Sn * Sn, (long long)Dn * Sn, (long long)Sn * Dn, 1.0f);

    // 7) out = x_attn @ WO^T -> fp32 x region of d_out
    gemm_bf16_wmma<false, true, false><<<gProj, blk, 0, stream>>>(
        xb, WOb, out_x, Bn * Sn, Dn, Dn, Dn, Dn, Dn, 0, 0, 0, 1.0f);
}
